// TMHA_39170101739876
// MI455X (gfx1250) — matrix operand order repaired and checked by reading, not compiled, not run on hardware
//
#include <hip/hip_runtime.h>
#include <hip/hip_bf16.h>

// ---------------------------------------------------------------------------
// Shapes (fixed by the reference): B=2, S=2048, DIM=512, H=8, HD=VD=64, QK=512
// ---------------------------------------------------------------------------
#define BB   2
#define SS   2048
#define DIMD 512
#define HH   8
#define HD   64
#define BS   (BB * SS)       // 4096 rows
#define TCOL 1536            // 2*QK + DIM

typedef __attribute__((ext_vector_type(16))) _Float16 v16h;
typedef __attribute__((ext_vector_type(8)))  float    v8f;

union U16h { v16h v; uint4 q[2]; };

// D = A(16x32 f16) * B(32x16 f16) + C(16x16 f32)
__device__ inline v8f wmma16(v16h a, v16h b, v8f c) {
  return __builtin_amdgcn_wmma_f32_16x16x32_f16(
      /*neg_a=*/false, a, /*neg_b=*/false, b,
      /*c_mod=*/(short)0, c, /*reuse_a=*/false, /*reuse_b=*/false);
}

// A-fragment (16x32, MxK) from row-major f16 [*, ld].
// Lane L: row = L&15; halves j=0..15 <- K = 16*(L>>4)+j.
__device__ inline v16h load_a(const _Float16* A, int ld, int row0, int k0, int lane) {
  const _Float16* p = A + (size_t)(row0 + (lane & 15)) * ld + k0 + 16 * (lane >> 4);
  U16h u;
  u.q[0] = *(const uint4*)(p);
  u.q[1] = *(const uint4*)(p + 8);
  return u.v;
}

// B-fragment (32x16, KxN) from "column-contiguous" storage Bc[n][k] row-major [*, ld].
// Lane L: col = L&15; halves j=0..15 <- K = 16*(L>>4)+j (32 contiguous bytes).
__device__ inline v16h load_b(const _Float16* Bc, int ld, int col0, int k0, int lane) {
  const _Float16* p = Bc + (size_t)(col0 + (lane & 15)) * ld + k0 + 16 * (lane >> 4);
  U16h u;
  u.q[0] = *(const uint4*)(p);
  u.q[1] = *(const uint4*)(p + 8);
  return u.v;
}

// Same B-fragment but from fp32 source, converting on load (rotator matrices).
__device__ inline v16h load_b_f32(const float* Bc, int ld, int col0, int k0, int lane) {
  const float* p = Bc + (size_t)(col0 + (lane & 15)) * ld + k0 + 16 * (lane >> 4);
  v16h r;
#pragma unroll
  for (int j = 0; j < 16; ++j) r[j] = (_Float16)p[j];
  return r;
}

// Async memory->LDS 16-byte transfer (GLOBAL_LOAD_ASYNC_TO_LDS_B128, ASYNCcnt).
// LDS byte offset = low 32 bits of the generic shared-space address.
__device__ inline void async_load_lds_b128(void* lds_dst, const void* gsrc) {
  unsigned int lds_off = (unsigned int)(size_t)lds_dst;
  asm volatile("global_load_async_to_lds_b128 %0, %1, off"
               :: "v"(lds_off), "v"(gsrc)
               : "memory");
}
__device__ inline void wait_asynccnt0() {
  asm volatile("s_wait_asynccnt 0x0" ::: "memory");
}
// Allow the 2 most recent async ops to stay outstanding (ping-pong prefetch).
__device__ inline void wait_asynccnt2() {
  asm volatile("s_wait_asynccnt 0x2" ::: "memory");
}

// ---------------------------------------------------------------------------
// fp32 -> f16 convert
// ---------------------------------------------------------------------------
__global__ void cvt_f32_f16(const float* __restrict__ src,
                            _Float16* __restrict__ dst, int n) {
  int i = blockIdx.x * blockDim.x + threadIdx.x;
  if (i < n) dst[i] = (_Float16)src[i];
}

// ---------------------------------------------------------------------------
// Wave-tile GEMM: C[M,N] = A[M,K] * Bc[N,K]^T  (torch-Linear layout).
// Each wave owns a 16x64 output tile (4 N-subtiles), K-loop of 16x16x32 WMMAs.
// The 8 waves of a WG cover consecutive M-tiles of the same N-block, so their
// B-fragment loads share cachelines.
// ---------------------------------------------------------------------------
template <bool F16OUT>
__global__ __launch_bounds__(256)
void gemm_wave(const _Float16* __restrict__ A, const _Float16* __restrict__ Bc,
               void* __restrict__ C, int M, int N, int K) {
  int lane = threadIdx.x & 31;
  int wid  = (blockIdx.x * blockDim.x + threadIdx.x) >> 5;
  int mtiles = M >> 4;
  int mt  = wid % mtiles;
  int nt4 = wid / mtiles;
  int row0 = mt << 4, col0 = nt4 << 6;
  if (col0 >= N) return;

  v8f acc[4] = {};
  for (int k0 = 0; k0 < K; k0 += 32) {
    v16h a = load_a(A, K, row0, k0, lane);
#pragma unroll
    for (int j = 0; j < 4; ++j) {
      v16h b = load_b(Bc, K, col0 + 16 * j, k0, lane);
      acc[j] = wmma16(a, b, acc[j]);
    }
  }
  // C layout: VGPR r -> row = row0 + r + 8*(lane>>4), col = col0 + 16*j + (lane&15)
  int g = lane >> 4, n = lane & 15;
#pragma unroll
  for (int j = 0; j < 4; ++j)
#pragma unroll
    for (int r = 0; r < 8; ++r) {
      size_t idx = (size_t)(row0 + r + 8 * g) * N + col0 + 16 * j + n;
      if (F16OUT) ((_Float16*)C)[idx] = (_Float16)acc[j][r];
      else        ((float*)C)[idx]    = acc[j][r];
    }
}

// ---------------------------------------------------------------------------
// Rotation + pack. One wave per position p=(b,s); mask[p] is WAVE-UNIFORM, so
// the rotate-vs-copy decision is a scalar branch (no exec divergence), and the
// unmasked path skips the 16KB rotator read entirely.
// Rotated path: stack 8 q-heads + 8 k-heads -> A (16x64); B = R^T whose column
// i is row i of the row-major rotator. Lanes 0-15 (g=0) hold q rows, lanes
// 16-31 (g=1) hold k rows -> single per-lane destination pointer select.
// Also scatters v into v^T [B*H][64][S] for the W@V WMMA.
// ---------------------------------------------------------------------------
__global__ __launch_bounds__(256)
void rotate_pack(const _Float16* __restrict__ th, const float* __restrict__ rot,
                 const int* __restrict__ mask,
                 _Float16* __restrict__ qg, _Float16* __restrict__ kg,
                 _Float16* __restrict__ vtg) {
  int lane = threadIdx.x & 31;
  int p = (blockIdx.x * blockDim.x + threadIdx.x) >> 5;  // [0, B*S)
  const _Float16* tp = th + (size_t)p * TCOL;
  int b = p >> 11;        // S = 2048
  int s = p & (SS - 1);

  int n = lane & 15, g = lane >> 4;
  // Per-lane dest: g=0 -> q heads, g=1 -> k heads; head index is the reg index r.
  _Float16* dst0 = g ? kg : qg;

  int msk = __builtin_amdgcn_readfirstlane(mask[p]);   // wave-uniform scalar
  if (msk) {
    const float* R = rot + (size_t)p * HD * HD;
    v16h a0 = load_a(tp, HD, 0, 0, lane);    // rows 0..15 = [q heads ; k heads]
    v16h a1 = load_a(tp, HD, 0, 32, lane);
#pragma unroll
    for (int nt = 0; nt < 4; ++nt) {
      v8f acc = {};
      v16h b0 = load_b_f32(R, HD, nt * 16, 0, lane);
      acc = wmma16(a0, b0, acc);
      v16h b1 = load_b_f32(R, HD, nt * 16, 32, lane);
      acc = wmma16(a1, b1, acc);
#pragma unroll
      for (int r = 0; r < 8; ++r)
        dst0[(((size_t)b * HH + r) * SS + s) * HD + nt * 16 + n] =
            (_Float16)acc[r];
    }
  } else {
    // Straight copy of q|k halves (1024 halves = 128 uint4, 4 per lane).
#pragma unroll
    for (int j = 0; j < 4; ++j) {
      int idx = (j * 32 + lane) * 8;        // 0..1016, step 8 halves (16B)
      int m = idx >> 6;                     // 0..15 (head row)
      int i = idx & 63;
      int hr = m & 7;
      _Float16* dq = (m < 8) ? qg : kg;
      *(uint4*)&dq[(((size_t)b * HH + hr) * SS + s) * HD + i] =
          *(const uint4*)&tp[idx];
    }
  }

  // v^T scatter: 512 elements / 32 lanes (common to both paths)
#pragma unroll
  for (int j = 0; j < 16; ++j) {
    int idx = lane * 16 + j;            // 0..511
    int h = idx >> 6, d = idx & 63;
    vtg[(((size_t)b * HH + h) * HD + d) * SS + s] = tp[2 * DIMD + idx];
  }
}

// ---------------------------------------------------------------------------
// Taylor attention (flash-style, no max trick needed: w = 1+s+s^2/2 > 0).
// WG = (b,h, 128 q-rows); 8 waves, each owning 16 q-rows. K/V staged in LDS
// in 32-key tiles via GLOBAL_LOAD_ASYNC_TO_LDS_B128 (ASYNCcnt), DOUBLE-
// BUFFERED: tile i+1 streams into the ping-pong buffer while tile i computes
// (s_wait_asynccnt 2 retires tile i's in-order ops; barrier makes it WG-wide).
// Probability tile goes through a per-wave LDS patch to convert C-layout f32
// -> A-layout f16 (in-wave DS ordering, no barrier).
// ---------------------------------------------------------------------------
__global__ __launch_bounds__(256)
void taylor_attn(const _Float16* __restrict__ qg, const _Float16* __restrict__ kg,
                 const _Float16* __restrict__ vtg, const int* __restrict__ mask,
                 _Float16* __restrict__ og) {
  __shared__ _Float16 ldsK[2][32 * 64];    // [buf][key][hd]
  __shared__ _Float16 ldsV[2][64 * 32];    // [buf][dim][key]
  __shared__ _Float16 ldsW[8][16 * 32];    // per-wave prob tile [qrow][key]

  int lane = threadIdx.x & 31;
  int wv   = threadIdx.x >> 5;             // 0..7
  int bh   = blockIdx.x >> 4;              // 0..15
  int qb   = blockIdx.x & 15;
  int b    = bh >> 3;
  int h    = bh & 7;

  const _Float16* Q  = qg  + (size_t)bh * SS * HD;
  const _Float16* Kp = kg  + (size_t)bh * SS * HD;
  const _Float16* Vt = vtg + (size_t)bh * HD * SS;
  const int* mrow = mask + b * SS;

  int row0 = qb * 128 + wv * 16;
  v16h qa0 = load_a(Q, HD, row0, 0, lane);
  v16h qa1 = load_a(Q, HD, row0, 32, lane);

  v8f o[4] = {};
  float wacc[8] = {};
  int t = threadIdx.x;
  int g = lane >> 4, n = lane & 15;

  // Per-thread staging coordinates (8 halves = 16B each)
  int idx = t * 8;
  int kr = idx >> 6, kc = idx & 63;        // K tile:   [32][64]
  int vr = idx >> 5, vc = idx & 31;        // V^T tile: [64][32]

  const int NT = SS / 32;                  // 64 key tiles
  // Prologue: stage tile 0 into buffer 0.
  async_load_lds_b128(&ldsK[0][idx], &Kp[(size_t)kr * HD + kc]);
  async_load_lds_b128(&ldsV[0][idx], &Vt[(size_t)vr * SS + vc]);

  for (int it = 0; it < NT; ++it) {
    int kb = it * 32;
    int cur = it & 1;
    if (it + 1 < NT) {
      int kb2 = kb + 32, nxt = (it + 1) & 1;
      async_load_lds_b128(&ldsK[nxt][idx], &Kp[(size_t)(kb2 + kr) * HD + kc]);
      async_load_lds_b128(&ldsV[nxt][idx], &Vt[(size_t)vr * SS + kb2 + vc]);
      wait_asynccnt2();                    // tile `it` landed (in-order)
    } else {
      wait_asynccnt0();
    }
    __syncthreads();                       // ... for every wave in the WG

    const _Float16* lK = &ldsK[cur][0];
    const _Float16* lV = &ldsV[cur][0];
#pragma unroll
    for (int sub = 0; sub < 2; ++sub) {
      v8f sacc = {};
      v16h b0 = load_b(lK, HD, sub * 16, 0, lane);
      sacc = wmma16(qa0, b0, sacc);
      v16h b1 = load_b(lK, HD, sub * 16, 32, lane);
      sacc = wmma16(qa1, b1, sacc);

      float mk = mrow[kb + sub * 16 + n] ? 1.f : 0.f;
      _Float16* wdst = &ldsW[wv][0];
#pragma unroll
      for (int r = 0; r < 8; ++r) {
        float s = sacc[r] * 0.125f;                 // 1/sqrt(HD)
        float w = mk * (1.f + s + 0.5f * s * s);
        wacc[r] += w;
        wdst[(r + 8 * g) * 32 + sub * 16 + n] = (_Float16)w;
      }
    }
    // In-wave DS store->load is in order; reload probs as an A-fragment.
    v16h wa = load_a(&ldsW[wv][0], 32, 0, 0, lane);
#pragma unroll
    for (int nt = 0; nt < 4; ++nt) {
      v16h vb = load_b(lV, 32, nt * 16, 0, lane);
      o[nt] = wmma16(wa, vb, o[nt]);
    }
    __syncthreads();  // all waves done reading buf `cur` before it is restaged
  }

  // Row sums: reduce each per-lane partial over the 16 lanes of its group.
#pragma unroll
  for (int r = 0; r < 8; ++r) {
    float v = wacc[r];
    v += __shfl_xor(v, 1, 32);
    v += __shfl_xor(v, 2, 32);
    v += __shfl_xor(v, 4, 32);
    v += __shfl_xor(v, 8, 32);
    wacc[r] = 1.f / v;
  }

#pragma unroll
  for (int nt = 0; nt < 4; ++nt)
#pragma unroll
    for (int r = 0; r < 8; ++r) {
      int row = row0 + r + 8 * g;
      int col = h * HD + nt * 16 + n;
      og[((size_t)(b * SS + row)) * DIMD + col] = (_Float16)(o[nt][r] * wacc[r]);
    }
}

// ---------------------------------------------------------------------------
// Launch
// ---------------------------------------------------------------------------
extern "C" void kernel_launch(void* const* d_in, const int* in_sizes, int n_in,
                              void* d_out, int out_size, void* d_ws, size_t ws_size,
                              hipStream_t stream) {
  const float* x    = (const float*)d_in[0];
  const int*   mask = (const int*)d_in[1];      // bool mask -> integer per harness
  const float* rot  = (const float*)d_in[2];
  const float* Wt   = (const float*)d_in[3];
  const float* Wo   = (const float*)d_in[4];

  char* ws = (char*)d_ws;
  size_t off = 0;
  auto alloc = [&](size_t bytes) {
    void* p = ws + off;
    off = (off + bytes + 255) & ~(size_t)255;
    return p;
  };
  _Float16* xh  = (_Float16*)alloc((size_t)BS * DIMD * 2);
  _Float16* wth = (_Float16*)alloc((size_t)TCOL * DIMD * 2);
  _Float16* woh = (_Float16*)alloc((size_t)DIMD * DIMD * 2);
  _Float16* th  = (_Float16*)alloc((size_t)BS * TCOL * 2);
  _Float16* qh  = (_Float16*)alloc((size_t)BS * DIMD * 2);
  _Float16* kh  = (_Float16*)alloc((size_t)BS * DIMD * 2);
  _Float16* vth = (_Float16*)alloc((size_t)BS * DIMD * 2);
  _Float16* ah  = (_Float16*)alloc((size_t)BS * DIMD * 2);
  (void)off; (void)ws_size; (void)in_sizes; (void)n_in; (void)out_size;

  cvt_f32_f16<<<(BS * DIMD) / 256, 256, 0, stream>>>(x, xh, BS * DIMD);
  cvt_f32_f16<<<(TCOL * DIMD) / 256, 256, 0, stream>>>(Wt, wth, TCOL * DIMD);
  cvt_f32_f16<<<(DIMD * DIMD) / 256, 256, 0, stream>>>(Wo, woh, DIMD * DIMD);

  // t = x @ W_trans^T : M=4096, N=1536, K=512 -> 256*24 waves -> 768 blocks
  gemm_wave<true><<<768, 256, 0, stream>>>(xh, wth, th, BS, TCOL, DIMD);

  // per-position rotation of q/k + v transpose pack: 1 wave / position
  rotate_pack<<<(BS * 32) / 256, 256, 0, stream>>>(th, rot, mask, qh, kh, vth);

  // attention: 16 (b,h) * 16 q-blocks = 256 WGs
  taylor_attn<<<256, 256, 0, stream>>>(qh, kh, vth, mask, ah);

  // out = attn @ W_o^T : M=4096, N=512, K=512 -> 256*8 waves -> 256 blocks
  gemm_wave<false><<<256, 256, 0, stream>>>(ah, woh, d_out, BS, DIMD, DIMD);
}
